// SwinTransformer_12962211299914
// MI455X (gfx1250) — compile-verified, matrix operand order checked by reading, not run on hardware
//
#include <hip/hip_runtime.h>
#include <hip/hip_bf16.h>

// ---------------------------------------------------------------------------
// Types for CDNA5 WMMA (wave32, 16x16x32 bf16 -> f32)
// ---------------------------------------------------------------------------
typedef __attribute__((ext_vector_type(16))) __bf16         v16bf;
typedef __attribute__((ext_vector_type(8)))  float          v8f;
typedef __attribute__((ext_vector_type(8)))  unsigned short v8us;

__device__ __forceinline__ unsigned short f2bf(float f) {
    unsigned int u = __float_as_uint(f);
    unsigned int r = u + 0x7FFFu + ((u >> 16) & 1u);   // round-to-nearest-even
    return (unsigned short)(r >> 16);
}

// CDNA5 async global->LDS copy (ASYNCcnt-tracked).  lds_off = low 32 bits of
// the generic shared-memory address (== wave-relative LDS byte offset).
__device__ __forceinline__ void async_copy_b128(unsigned lds_off, const void* g) {
    asm volatile("global_load_async_to_lds_b128 %0, %1, off"
                 :: "v"(lds_off), "v"(g) : "memory");
}
__device__ __forceinline__ void wait_async0() {
    asm volatile("s_wait_asynccnt 0x0" ::: "memory");
}

// Load one 16x16x32 bf16 fragment from a bf16 LDS tile (row stride 32).
// lane&15 selects the M (or N) row; lane>>4 selects the K-half per the ISA
// layout: lanes 0-15 : K = {0..7, 16..23}, lanes 16-31 : K = {8..15, 24..31}.
__device__ __forceinline__ v16bf load_frag(const unsigned short* rowbase, int lhi) {
    union { v16bf bf; v8us h[2]; } u;
    u.h[0] = *reinterpret_cast<const v8us*>(rowbase + lhi * 8);
    u.h[1] = *reinterpret_cast<const v8us*>(rowbase + 16 + lhi * 8);
    return u.bf;
}

// f32 -> packed bf16 pre-pass (2 floats -> 1 dword)
__global__ void cvt_bf16_kernel(const float* __restrict__ src,
                                unsigned int* __restrict__ dst, long npair)
{
    long i = blockIdx.x * (long)blockDim.x + threadIdx.x;
    if (i >= npair) return;
    float a = src[2 * i], b = src[2 * i + 1];
    dst[i] = (unsigned)f2bf(a) | ((unsigned)f2bf(b) << 16);
}

// ---------------------------------------------------------------------------
// GEMM:  C[M,N] = act(Abf[M,K] @ Wbf[N,K]^T + bias) (+ resid)
// BM=64, BN=128, BK=32; 256 threads = 8 waves, each wave -> 32x32 (2x2 WMMA).
// bf16 tiles double-buffered in LDS (24 KB), filled by async global->LDS
// copies when fully in-bounds, guarded synchronous fill otherwise.
// ---------------------------------------------------------------------------
#define BM 64
#define BN 128
#define BK 32

#define A_TILE (BM * BK)            // 2048 bf16
#define B_TILE (BN * BK)            // 4096 bf16
#define B_BASE (2 * A_TILE)         // B buffers start after the two A buffers

__device__ __forceinline__ void stage_tileA(unsigned short* dst,
                                            const unsigned short* __restrict__ A,
                                            int bm, int k0, int M, int K, int tid)
{
    if ((bm + BM <= M) && (k0 + BK <= K)) {
        unsigned base = (unsigned)(size_t)dst;
        int i = tid;                                   // 256 x 16B chunks (8 bf16)
        const unsigned short* g = A + (size_t)(bm + (i >> 2)) * K + (k0 + ((i & 3) << 3));
        async_copy_b128(base + (unsigned)i * 16u, g);
    } else {
        for (int i = tid; i < BM * BK; i += 256) {
            int r = i >> 5, c = i & 31;
            int gm = bm + r, gk = k0 + c;
            dst[i] = (gm < M && gk < K) ? A[(size_t)gm * K + gk] : (unsigned short)0;
        }
    }
}

__device__ __forceinline__ void stage_tileB(unsigned short* dst,
                                            const unsigned short* __restrict__ W,
                                            int bn, int k0, int N, int K, int tid)
{
    if ((bn + BN <= N) && (k0 + BK <= K)) {
        unsigned base = (unsigned)(size_t)dst;
#pragma unroll
        for (int ii = 0; ii < 2; ++ii) {               // 512 x 16B chunks
            int i = tid + ii * 256;
            const unsigned short* g = W + (size_t)(bn + (i >> 2)) * K + (k0 + ((i & 3) << 3));
            async_copy_b128(base + (unsigned)i * 16u, g);
        }
    } else {
        for (int i = tid; i < BN * BK; i += 256) {
            int r = i >> 5, c = i & 31;
            int gn = bn + r, gk = k0 + c;
            dst[i] = (gn < N && gk < K) ? W[(size_t)gn * K + gk] : (unsigned short)0;
        }
    }
}

__global__ __launch_bounds__(256) void gemm_bf16_kernel(
    const unsigned short* __restrict__ A, const unsigned short* __restrict__ W,
    const float* __restrict__ bias, const float* __restrict__ resid,
    float* __restrict__ C, int M, int N, int K, int act)
{
    // [A0 | A1 | B0 | B1] as bf16: (2048 + 2048 + 4096 + 4096) * 2B = 24 KB
    __shared__ __align__(16) unsigned short sMem[2 * A_TILE + 2 * B_TILE];

    const int tid  = threadIdx.x;
    const int bm   = blockIdx.y * BM;
    const int bn   = blockIdx.x * BN;
    const int wave = tid >> 5;
    const int lane = tid & 31;
    const int wm   = wave & 1;       // 2 M-strips of 32
    const int wn   = wave >> 1;      // 4 N-strips of 32
    const int lr   = lane & 15;
    const int lhi  = lane >> 4;

    v8f acc[2][2];
    for (int mi = 0; mi < 2; ++mi)
        for (int ni = 0; ni < 2; ++ni)
            for (int r = 0; r < 8; ++r) acc[mi][ni][r] = 0.0f;

    const int nk = (K + BK - 1) / BK;

    // Prologue: fill buffer 0
    stage_tileA(sMem, A, bm, 0, M, K, tid);
    stage_tileB(sMem + B_BASE, W, bn, 0, N, K, tid);

    for (int kt = 0; kt < nk; ++kt) {
        wait_async0();          // my async writes into buffer (kt&1) are done
        __syncthreads();        // everyone's writes (async or ds) are visible

        if (kt + 1 < nk) {      // overlap: fill the other buffer while computing
            int nb = (kt + 1) & 1;
            stage_tileA(sMem + nb * A_TILE, A, bm, (kt + 1) * BK, M, K, tid);
            stage_tileB(sMem + B_BASE + nb * B_TILE, W, bn, (kt + 1) * BK, N, K, tid);
        }

        const unsigned short* aT = sMem + (kt & 1) * A_TILE;
        const unsigned short* bT = sMem + B_BASE + (kt & 1) * B_TILE;

        v16bf af[2], bq[2];
        for (int mi = 0; mi < 2; ++mi)
            af[mi] = load_frag(aT + (wm * 32 + mi * 16 + lr) * BK, lhi);
        for (int ni = 0; ni < 2; ++ni)
            bq[ni] = load_frag(bT + (wn * 32 + ni * 16 + lr) * BK, lhi);

        for (int mi = 0; mi < 2; ++mi)
            for (int ni = 0; ni < 2; ++ni)
                acc[mi][ni] = __builtin_amdgcn_wmma_f32_16x16x32_bf16(
                    false, af[mi], false, bq[ni], (short)0, acc[mi][ni],
                    false, false);
        // next iteration's head barrier protects buffer (kt&1) from refill
    }

    for (int mi = 0; mi < 2; ++mi) {
        for (int ni = 0; ni < 2; ++ni) {
            int col = bn + wn * 32 + ni * 16 + lr;
            for (int r = 0; r < 8; ++r) {
                int row = bm + wm * 32 + mi * 16 + r + 8 * lhi;
                if (row < M && col < N) {
                    float v = acc[mi][ni][r];
                    if (bias)  v += bias[col];
                    if (act == 1) v = v / (1.0f + __expf(-v));   // SiLU
                    if (resid) v += resid[(size_t)row * N + col];
                    C[(size_t)row * N + col] = v;
                }
            }
        }
    }
}

// ---------------------------------------------------------------------------
// LayerNorm over last dim (block per row), optional fused SiLU
// ---------------------------------------------------------------------------
__global__ __launch_bounds__(256) void ln_kernel(
    const float* __restrict__ x, float* __restrict__ y,
    const float* __restrict__ g, const float* __restrict__ b,
    int D, float eps, int silu)
{
    __shared__ float sbuf[256];
    const size_t row = blockIdx.x;
    const float* xr = x + row * (size_t)D;
    float* yr = y + row * (size_t)D;

    float s = 0.0f;
    for (int i = threadIdx.x; i < D; i += 256) s += xr[i];
    sbuf[threadIdx.x] = s; __syncthreads();
    for (int o = 128; o > 0; o >>= 1) {
        if (threadIdx.x < o) sbuf[threadIdx.x] += sbuf[threadIdx.x + o];
        __syncthreads();
    }
    float mu = sbuf[0] / (float)D; __syncthreads();

    float v = 0.0f;
    for (int i = threadIdx.x; i < D; i += 256) { float t = xr[i] - mu; v += t * t; }
    sbuf[threadIdx.x] = v; __syncthreads();
    for (int o = 128; o > 0; o >>= 1) {
        if (threadIdx.x < o) sbuf[threadIdx.x] += sbuf[threadIdx.x + o];
        __syncthreads();
    }
    float rstd = rsqrtf(sbuf[0] / (float)D + eps);

    for (int i = threadIdx.x; i < D; i += 256) {
        float val = (xr[i] - mu) * rstd * g[i] + b[i];
        if (silu) val = val / (1.0f + __expf(-val));
        yr[i] = val;
    }
}

// ---------------------------------------------------------------------------
// Patch-merging rearrange: [B, Hs*red, Ws*red, ind] -> [B*Hs*Ws, red*red*ind]
// (stage 0 reads NCHW input)
// ---------------------------------------------------------------------------
__global__ void patch_kernel(const float* __restrict__ src, float* __restrict__ dst,
                             int Hs, int Ws, int red, int ind, int nchw,
                             int Hin, int Win, long total)
{
    long e = blockIdx.x * (long)blockDim.x + threadIdx.x;
    if (e >= total) return;
    long t = e;
    int c  = t % ind; t /= ind;
    int rx = t % red; t /= red;
    int ry = t % red; t /= red;
    int wx = t % Ws;  t /= Ws;
    int hy = t % Hs;  int b = (int)(t / Hs);
    int y = hy * red + ry, x = wx * red + rx;
    float v;
    if (nchw) v = src[((long)(b * ind + c) * Hin + y) * Win + x];
    else      v = src[((long)(b * Hin + y) * Win + x) * ind + c];
    dst[e] = v;
}

// ---------------------------------------------------------------------------
// Cyclic shift (roll by (-3,-3)): dst[y,x] = src[(y+3)%H, (x+3)%W]
// ---------------------------------------------------------------------------
__global__ void roll_kernel(const float* __restrict__ src, float* __restrict__ dst,
                            int Hp, int Wp, int d, long total)
{
    long i = blockIdx.x * (long)blockDim.x + threadIdx.x;
    if (i >= total) return;
    int c = i % d; long t = i / d;
    int x = t % Wp; t /= Wp;
    int y = t % Hp; long b = t / Hp;
    int ys = (y + 3) % Hp, xs = (x + 3) % Wp;
    dst[i] = src[((b * Hp + ys) * (long)Wp + xs) * d + c];
}

// out[y,x] += shift ? tmp[(y-3)%H, (x-3)%W] : tmp[y,x]
__global__ void addroll_kernel(float* __restrict__ out, const float* __restrict__ tmp,
                               int Hp, int Wp, int d, int shift, long total)
{
    long i = blockIdx.x * (long)blockDim.x + threadIdx.x;
    if (i >= total) return;
    if (!shift) { out[i] += tmp[i]; return; }
    int c = i % d; long t = i / d;
    int x = t % Wp; t /= Wp;
    int y = t % Hp; long b = t / Hp;
    int ys = (y - 3 + Hp) % Hp, xs = (x - 3 + Wp) % Wp;
    out[i] += tmp[((b * Hp + ys) * (long)Wp + xs) * d + c];
}

// ---------------------------------------------------------------------------
// Windowed attention: one block per (window, head, batch); ws=7, dh=32.
// Relative-position index & shift mask computed analytically (matches
// make_pos_mask): pos = (dy+6)*13 + (dx+6) with dy = yJ - yI in rolled coords;
// masked when |dy|>=7 or |dx|>=7.
// ---------------------------------------------------------------------------
__global__ __launch_bounds__(64) void attn_kernel(
    const float* __restrict__ qkv, float* __restrict__ obuf,
    const float* __restrict__ rel, int Hp, int Wp, int d, int nh, int shift)
{
    const int win  = blockIdx.x;
    const int head = blockIdx.y;
    const int b    = blockIdx.z;
    const int wstr = Wp / 7;
    const int wh = win / wstr, ww = win % wstr;
    const int d3 = 3 * d;

    __shared__ float sk[49 * 32];
    __shared__ float sv[49 * 32];

    const int tid = threadIdx.x;
    for (int i = tid; i < 49 * 32; i += 64) {
        int j = i >> 5, t = i & 31;
        int gy = wh * 7 + (j / 7), gx = ww * 7 + (j % 7);
        size_t row = ((size_t)(b * Hp + gy)) * Wp + gx;
        const float* base = qkv + row * d3 + head * 32 + t;
        sk[i] = base[d];
        sv[i] = base[2 * d];
    }
    __syncthreads();

    const int i = tid;
    if (i < 49) {
        const int ri = i / 7, ci = i % 7;
        float q[32];
        {
            int gy = wh * 7 + ri, gx = ww * 7 + ci;
            size_t row = ((size_t)(b * Hp + gy)) * Wp + gx;
            const float* qp = qkv + row * d3 + head * 32;
            for (int t = 0; t < 32; ++t) q[t] = qp[t];
        }
        int yI = 0, xI = 0;
        if (shift) { yI = (wh * 7 + ri + 3) % Hp; xI = (ww * 7 + ci + 3) % Wp; }

        const float scale = 0.17677669529663687f;  // 1/sqrt(32)
        float sc[49];
        float mx = -__builtin_inff();
        for (int j = 0; j < 49; ++j) {
            float s = 0.0f;
            for (int t = 0; t < 32; ++t) s += q[t] * sk[j * 32 + t];
            s *= scale;
            int rj = j / 7, cj = j % 7;
            int dy, dx; bool ok = true;
            if (shift) {
                int yJ = (wh * 7 + rj + 3) % Hp, xJ = (ww * 7 + cj + 3) % Wp;
                dy = yJ - yI; dx = xJ - xI;
                ok = (dy < 7 && dy > -7 && dx < 7 && dx > -7);
                if (!ok) { dy = 0; dx = 0; }
            } else { dy = rj - ri; dx = cj - ci; }
            int pos = (dy + 6) * 13 + (dx + 6);
            s += rel[pos * nh + head];
            sc[j] = ok ? s : -__builtin_inff();
            mx = fmaxf(mx, sc[j]);
        }
        float sum = 0.0f;
        for (int j = 0; j < 49; ++j) { float e = __expf(sc[j] - mx); sc[j] = e; sum += e; }
        float inv = 1.0f / sum;

        float o[32];
        for (int t = 0; t < 32; ++t) o[t] = 0.0f;
        for (int j = 0; j < 49; ++j) {
            float a = sc[j] * inv;
            for (int t = 0; t < 32; ++t) o[t] += a * sv[j * 32 + t];
        }
        int gy = wh * 7 + ri, gx = ww * 7 + ci;
        size_t row = ((size_t)(b * Hp + gy)) * Wp + gx;
        float* op = obuf + row * d + head * 32;
        for (int t = 0; t < 32; ++t) op[t] = o[t];
    }
}

// mean over tokens: src[B, ntok, D] -> dst[B, D]
__global__ void meanpool_kernel(const float* __restrict__ src, float* __restrict__ dst,
                                int ntok, int D, int total)
{
    int i = blockIdx.x * blockDim.x + threadIdx.x;
    if (i >= total) return;
    int c = i % D; int b = i / D;
    float s = 0.0f;
    for (int t = 0; t < ntok; ++t) s += src[((long)b * ntok + t) * D + c];
    dst[i] = s / (float)ntok;
}

// ---------------------------------------------------------------------------
// Host orchestration
// ---------------------------------------------------------------------------
static inline long cdiv(long a, long b) { return (a + b - 1) / b; }

// convert A and W to bf16 scratch, then run the WMMA GEMM
static inline void launch_gemm(const float* A, const float* W, const float* bias,
                               const float* resid, float* C, int M, int N, int K,
                               int act, unsigned short* ABF, unsigned short* WBF,
                               hipStream_t s)
{
    long na = (long)M * K / 2, nw = (long)N * K / 2;   // all M*K, N*K are even
    cvt_bf16_kernel<<<(unsigned)cdiv(na, 256), 256, 0, s>>>(A, (unsigned int*)ABF, na);
    cvt_bf16_kernel<<<(unsigned)cdiv(nw, 256), 256, 0, s>>>(W, (unsigned int*)WBF, nw);
    dim3 g((unsigned)cdiv(N, BN), (unsigned)cdiv(M, BM));
    gemm_bf16_kernel<<<g, dim3(256), 0, s>>>(ABF, WBF, bias, resid, C, M, N, K, act);
}

extern "C" void kernel_launch(void* const* d_in, const int* in_sizes, int n_in,
                              void* d_out, int out_size, void* d_ws, size_t ws_size,
                              hipStream_t stream)
{
    (void)in_sizes; (void)n_in; (void)out_size; (void)ws_size;
    int idx = 0;
    auto nxt = [&]() -> const float* { return (const float*)d_in[idx++]; };

    const float* X = nxt();  // [8,3,224,224]

    static const int DIMS_[4]  = {96, 192, 384, 768};
    static const int NH_[4]    = {3, 6, 12, 24};
    static const int DEPTH_[4] = {2, 2, 6, 2};
    static const int DFF_[4]   = {384, 768, 1536, 3072};
    static const int RED_[4]   = {4, 2, 2, 2};
    static const int INDIM_[4] = {3, 96, 192, 384};
    const int B_ = 8;

    // Workspace layout
    const size_t SZ_OUT = 25088ull * 96;    // 2,408,448 floats
    const size_t SZ_QKV = 25088ull * 288;   // 7,225,344 floats
    const size_t SZ_MID = 25088ull * 384;   // 9,633,792 floats
    float* OUTb = (float*)d_ws;
    float* Hb   = OUTb + SZ_OUT;
    float* RL   = Hb   + SZ_OUT;
    float* QKVb = RL   + SZ_OUT;
    float* ATb  = QKVb + SZ_QKV;
    float* TMPb = ATb  + SZ_OUT;
    float* MIDb = TMPb + SZ_OUT;
    // bf16 scratch (16B-aligned: all preceding sizes are multiples of 4 floats)
    unsigned short* ABF = (unsigned short*)(MIDb + SZ_MID);   // max M*K = 9,633,792
    unsigned short* WBF = ABF + 9633792ull;                   // max N*K = 2,359,296

    int Hin = 224;  // spatial size entering the stage
    for (int s = 0; s < 4; ++s) {
        const float* ew    = nxt();
        const float* eb    = nxt();
        const float* eg    = nxt();
        const float* ebeta = nxt();
        const int red = RED_[s], ind = INDIM_[s], d = DIMS_[s];
        const int Hs = Hin / red;
        const int T  = B_ * Hs * Hs;
        const int Kp = red * red * ind;

        // patch rearrange -> TMPb [T, Kp]
        {
            long total = (long)T * Kp;
            const float* src = (s == 0) ? X : OUTb;
            patch_kernel<<<(unsigned)cdiv(total, 256), 256, 0, stream>>>(
                src, TMPb, Hs, Hs, red, ind, (s == 0) ? 1 : 0, (s == 0) ? 224 : Hin,
                (s == 0) ? 224 : Hin, total);
        }
        // embed linear + bias -> Hb, then LN(1e-5) -> OUTb
        launch_gemm(TMPb, ew, eb, nullptr, Hb, T, d, Kp, 0, ABF, WBF, stream);
        ln_kernel<<<T, 256, 0, stream>>>(Hb, OUTb, eg, ebeta, d, 1e-5f, 0);

        const long tot_td = (long)T * d;
        for (int l = 0; l < DEPTH_[s]; ++l) {
            const float* ng1  = nxt();
            const float* nb1  = nxt();
            const float* wqkv = nxt();
            const float* wo   = nxt();
            const float* bo   = nxt();
            const float* rel  = nxt();
            const float* ng2  = nxt();
            const float* nb2  = nxt();
            const float* w1   = nxt();
            const float* b1   = nxt();
            const float* w2   = nxt();
            const float* b2   = nxt();
            const int shift = (l % 2 == 0) ? 1 : 0;

            // h = LN(out)
            ln_kernel<<<T, 256, 0, stream>>>(OUTb, Hb, ng1, nb1, d, 1e-6f, 0);
            const float* attin = Hb;
            if (shift) {
                roll_kernel<<<(unsigned)cdiv(tot_td, 256), 256, 0, stream>>>(
                    Hb, RL, Hs, Hs, d, tot_td);
                attin = RL;
            }
            // qkv (no bias)
            launch_gemm(attin, wqkv, nullptr, nullptr, QKVb, T, 3 * d, d, 0,
                        ABF, WBF, stream);
            // attention
            {
                dim3 ag((unsigned)((Hs / 7) * (Hs / 7)), (unsigned)NH_[s], (unsigned)B_);
                attn_kernel<<<ag, 64, 0, stream>>>(QKVb, ATb, rel, Hs, Hs, d, NH_[s], shift);
            }
            // proj + bias
            launch_gemm(ATb, wo, bo, nullptr, TMPb, T, d, d, 0, ABF, WBF, stream);
            // out += roll-back(proj)
            addroll_kernel<<<(unsigned)cdiv(tot_td, 256), 256, 0, stream>>>(
                OUTb, TMPb, Hs, Hs, d, shift, tot_td);
            // FFN
            ln_kernel<<<T, 256, 0, stream>>>(OUTb, Hb, ng2, nb2, d, 1e-6f, 0);
            launch_gemm(Hb, w1, b1, nullptr, MIDb, T, DFF_[s], d, 1,
                        ABF, WBF, stream);                                   // SiLU
            launch_gemm(MIDb, w2, b2, OUTb, OUTb, T, d, DFF_[s], 0,
                        ABF, WBF, stream);                                   // +resid
        }
        Hin = Hs;
    }

    // Final head
    const float* g1  = nxt();
    const float* fb1 = nxt();
    const float* fw  = nxt();
    const float* fbb = nxt();
    const float* g2  = nxt();
    const float* fb2 = nxt();
    const float* cw  = nxt();
    const float* cb  = nxt();

    const int T3 = B_ * 49;
    ln_kernel<<<T3, 256, 0, stream>>>(OUTb, Hb, g1, fb1, 768, 1e-5f, 0);
    launch_gemm(Hb, fw, fbb, nullptr, TMPb, T3, 1536, 768, 0, ABF, WBF, stream);
    ln_kernel<<<T3, 256, 0, stream>>>(TMPb, MIDb, g2, fb2, 1536, 1e-5f, 1);  // LN + SiLU
    {
        int total = B_ * 1536;
        meanpool_kernel<<<(unsigned)cdiv(total, 256), 256, 0, stream>>>(
            MIDb, RL, 49, 1536, total);
    }
    launch_gemm(RL, cw, cb, nullptr, (float*)d_out, B_, 1000, 1536, 0, ABF, WBF, stream);
}
